// StepModelNoise_12506944766384
// MI455X (gfx1250) — matrix operand order repaired and checked by reading, not compiled, not run on hardware
//
#include <hip/hip_runtime.h>
#include <cstdint>
#include <cstddef>

#define BATCH 32
#define NPTS  2048

typedef _Float16 h16;
typedef __attribute__((ext_vector_type(2)))  _Float16 v2h;
typedef __attribute__((ext_vector_type(16))) _Float16 v16h;
typedef __attribute__((ext_vector_type(8)))  float    v8f;

enum { ACT_NONE = 0, ACT_RELU = 1, ACT_SIGMOID = 2 };

// PK2 layout for f16 activation tensors (B, C, P):
//   addr(b,c,p) = ((b*Chalf + c/2)*P + p)*2 + (c&1),  Chalf = ceil(C/2)
// -> a WMMA B-operand lane reads its 16 halves as 8 dwords (channel pairs), P dwords apart.
__device__ __forceinline__ size_t pk2i(int b, int c, int p, int Chalf, int P) {
  return (((size_t)b * Chalf + (c >> 1)) * P + p) * 2 + (c & 1);
}

// =====================================================================
// GEMM via CDNA5 WMMA:  Y[b,o,p] = sum_c W[o,c]*X[b,c,p] + bias[o]
// X: f16 PK2, CinP padded-even channels (pad channels zero)
// W: f16, rows padded+zeroed to CinA = roundup32(Cin) -> all A loads
//    unconditional & contiguous (merge into wide loads, no clamps)
// One wave computes a 16 x 64 output strip (4 WMMAs per 32-deep K step).
// =====================================================================
__global__ __launch_bounds__(32)
void gemm_wmma_kernel(const h16* __restrict__ X, const h16* __restrict__ W16,
                      const float* __restrict__ bias, float* __restrict__ Y,
                      int CinP, int CinA, int Cout, int P) {
  const int lane = threadIdx.x;
  const int n0 = blockIdx.x * 64;
  const int m0 = blockIdx.y * 16;
  const int b  = blockIdx.z;
  // pair-indexed view of this batch's activations, at this lane's column
  const v2h* __restrict__ Xq =
      (const v2h*)X + (size_t)b * (CinP >> 1) * P + n0 + (lane & 15);

  const int mrow = min(m0 + (lane & 15), Cout - 1);  // clamped A row
  const h16* __restrict__ Wr = W16 + (size_t)mrow * CinA;
  const int hoff = (lane < 16) ? 0 : 8;   // D row half
  const int boff = (lane < 16) ? 0 : 16;  // B K-offset half

  v8f acc0 = {}, acc1 = {}, acc2 = {}, acc3 = {};
  const int kfull = CinP & ~31;
  int k0 = 0;
#pragma unroll 2
  for (; k0 < kfull; k0 += 32) {
    v16h a, bm0, bm1, bm2, bm3;
#pragma unroll
    for (int h = 0; h < 16; ++h) {  // one contiguous 16-half run per lane
      int k = k0 + boff + h;
      a[h] = Wr[k];
    }
    const int kb = (k0 + boff) >> 1;
#pragma unroll
    for (int j = 0; j < 8; ++j) {   // one dword per pair of halves
      const v2h* xp = Xq + (size_t)(kb + j) * P;
      v2h p0 = xp[0], p1 = xp[16], p2 = xp[32], p3 = xp[48];
      bm0[2 * j] = p0.x; bm0[2 * j + 1] = p0.y;
      bm1[2 * j] = p1.x; bm1[2 * j + 1] = p1.y;
      bm2[2 * j] = p2.x; bm2[2 * j + 1] = p2.y;
      bm3[2 * j] = p3.x; bm3[2 * j + 1] = p3.y;
    }
    acc0 = __builtin_amdgcn_wmma_f32_16x16x32_f16(false, a, false, bm0, (short)0, acc0, false, false);
    acc1 = __builtin_amdgcn_wmma_f32_16x16x32_f16(false, a, false, bm1, (short)0, acc1, false, false);
    acc2 = __builtin_amdgcn_wmma_f32_16x16x32_f16(false, a, false, bm2, (short)0, acc2, false, false);
    acc3 = __builtin_amdgcn_wmma_f32_16x16x32_f16(false, a, false, bm3, (short)0, acc3, false, false);
  }
  if (k0 < CinP) {  // ragged tail (CinP%32 != 0); A is pad-safe up to CinA
    v16h a, bm0, bm1, bm2, bm3;
#pragma unroll
    for (int h = 0; h < 16; ++h) {
      int k = k0 + boff + h;
      a[h] = Wr[k];  // k < CinA always; pad region is zero
    }
#pragma unroll
    for (int j = 0; j < 8; ++j) {   // clamp + select, no divergent loads
      int c = k0 + boff + 2 * j;
      const v2h* xp = Xq + (size_t)(min(c, CinP - 2) >> 1) * P;
      v2h p0 = xp[0], p1 = xp[16], p2 = xp[32], p3 = xp[48];
      bool ok = c < CinP;
      h16 z = (h16)0;
      bm0[2 * j] = ok ? p0.x : z; bm0[2 * j + 1] = ok ? p0.y : z;
      bm1[2 * j] = ok ? p1.x : z; bm1[2 * j + 1] = ok ? p1.y : z;
      bm2[2 * j] = ok ? p2.x : z; bm2[2 * j + 1] = ok ? p2.y : z;
      bm3[2 * j] = ok ? p3.x : z; bm3[2 * j + 1] = ok ? p3.y : z;
    }
    acc0 = __builtin_amdgcn_wmma_f32_16x16x32_f16(false, a, false, bm0, (short)0, acc0, false, false);
    acc1 = __builtin_amdgcn_wmma_f32_16x16x32_f16(false, a, false, bm1, (short)0, acc1, false, false);
    acc2 = __builtin_amdgcn_wmma_f32_16x16x32_f16(false, a, false, bm2, (short)0, acc2, false, false);
    acc3 = __builtin_amdgcn_wmma_f32_16x16x32_f16(false, a, false, bm3, (short)0, acc3, false, false);
  }
  // D: M = r + (lane<16?0:8), N = lane%16 (+16/32/48 per tile); f32 plain
#pragma unroll
  for (int r = 0; r < 8; ++r) {
    int m = m0 + r + hoff;
    if (m < Cout) {
      float bb = bias[m];
      float* yp = Y + ((size_t)b * Cout + m) * P + n0 + (lane & 15);
      yp[0]  = acc0[r] + bb;
      yp[16] = acc1[r] + bb;
      yp[32] = acc2[r] + bb;
      yp[48] = acc3[r] + bb;
    }
  }
}

// weights f32 (Cout,Cin) -> f16 padded rows (Cout,CinA), pad = 0
__global__ void cvt16pad_kernel(const float* __restrict__ src, h16* __restrict__ dst,
                                int Cin, int CinA, size_t total) {
  size_t t = (size_t)blockIdx.x * blockDim.x + threadIdx.x;
  if (t >= total) return;
  int c = (int)(t % (size_t)CinA);
  int o = (int)(t / (size_t)CinA);
  dst[t] = (c < Cin) ? (h16)src[(size_t)o * Cin + c] : (h16)0;
}

// ============================ BatchNorm ==============================
__global__ void bn_stats_kernel(const float* __restrict__ Y,
                                float* __restrict__ sc, float* __restrict__ sh,
                                const float* __restrict__ g, const float* __restrict__ be,
                                int C, int S) {
  const int c = blockIdx.x;
  const int n = BATCH * S;
  float s = 0.0f, s2 = 0.0f;
  for (int i = threadIdx.x; i < n; i += blockDim.x) {
    int b = i / S, sp = i - b * S;
    float v = Y[((size_t)b * C + c) * S + sp];
    s += v; s2 += v * v;
  }
  __shared__ float rs[256], rs2[256];
  rs[threadIdx.x] = s; rs2[threadIdx.x] = s2;
  __syncthreads();
  for (int st = 128; st > 0; st >>= 1) {
    if (threadIdx.x < st) { rs[threadIdx.x] += rs[threadIdx.x + st];
                            rs2[threadIdx.x] += rs2[threadIdx.x + st]; }
    __syncthreads();
  }
  if (threadIdx.x == 0) {
    float mean = rs[0] / (float)n;
    float var  = rs2[0] / (float)n - mean * mean;
    float scv  = g[c] * rsqrtf(var + 1e-5f);
    sc[c] = scv;
    sh[c] = be[c] - mean * scv;
  }
}

// f32 pre-BN (plain) -> normalized+activated f16 (PK2). C is even.
// t IS the PK2 linear address -> coalesced writes.
__global__ void bn_apply_kernel(const float* __restrict__ Y, h16* __restrict__ out,
                                const float* __restrict__ sc, const float* __restrict__ sh,
                                int C, int S, int act, size_t total) {
  size_t t = (size_t)blockIdx.x * blockDim.x + threadIdx.x;
  if (t >= total) return;
  int sub = (int)(t & 1);
  size_t q = t >> 1;
  int p  = (int)(q % (size_t)S);
  int cp = (int)((q / (size_t)S) % (size_t)(C >> 1));
  int b  = (int)(q / ((size_t)S * (C >> 1)));
  int c  = cp * 2 + sub;
  float v = Y[((size_t)b * C + c) * S + p] * sc[c] + sh[c];
  if (act == ACT_RELU)         v = fmaxf(v, 0.0f);
  else if (act == ACT_SIGMOID) v = 1.0f / (1.0f + expf(-v));
  out[t] = (h16)v;
}

// ===================== Farthest point sampling =======================
__global__ void fps_kernel(const float* __restrict__ xyz, int* __restrict__ idx,
                           int N, int npoint) {
  const int b = blockIdx.x;
  const float* X = xyz + (size_t)b * 3 * N;
  __shared__ float dist[NPTS];
  __shared__ float rv[256];
  __shared__ int   ri[256];
  __shared__ int   s_far;
  const int tid = threadIdx.x;
  for (int i = tid; i < N; i += blockDim.x) dist[i] = 1e10f;
  if (tid == 0) s_far = 0;
  __syncthreads();
  for (int it = 0; it < npoint; ++it) {
    int far = s_far;
    if (tid == 0) idx[b * npoint + it] = far;
    float fx = X[far], fy = X[N + far], fz = X[2 * N + far];
    float bv = -1.0f; int bi = 0;
    for (int i = tid; i < N; i += blockDim.x) {
      float dx = X[i] - fx, dy = X[N + i] - fy, dz = X[2 * N + i] - fz;
      float nd = fminf(dist[i], dx * dx + dy * dy + dz * dz);
      dist[i] = nd;
      if (nd > bv) { bv = nd; bi = i; }
    }
    rv[tid] = bv; ri[tid] = bi;
    __syncthreads();
    for (int st = blockDim.x / 2; st > 0; st >>= 1) {
      if (tid < st) {
        if (rv[tid + st] > rv[tid] ||
            (rv[tid + st] == rv[tid] && ri[tid + st] < ri[tid])) {
          rv[tid] = rv[tid + st]; ri[tid] = ri[tid + st];
        }
      }
      __syncthreads();
    }
    if (tid == 0) s_far = ri[0];
    __syncthreads();
  }
}

__global__ void gather_xyz_kernel(const float* __restrict__ xyz, const int* __restrict__ idx,
                                  float* __restrict__ out, int N, int S, size_t total) {
  size_t t = (size_t)blockIdx.x * blockDim.x + threadIdx.x;
  if (t >= total) return;
  int s = (int)(t % (size_t)S);
  int d = (int)((t / (size_t)S) % 3);
  int b = (int)(t / ((size_t)3 * S));
  out[t] = xyz[((size_t)b * 3 + d) * N + idx[b * S + s]];
}

// ============================ Ball query =============================
__global__ void ball_query_kernel(const float* __restrict__ xyz,
                                  const float* __restrict__ newxyz,
                                  int* __restrict__ out,
                                  int N, int S, int K, float r2, size_t total) {
  size_t t = (size_t)blockIdx.x * blockDim.x + threadIdx.x;
  if (t >= total) return;
  int b = (int)(t / (size_t)S), s = (int)(t % (size_t)S);
  const float* X = xyz + (size_t)b * 3 * N;
  float qx = newxyz[((size_t)b * 3 + 0) * S + s];
  float qy = newxyz[((size_t)b * 3 + 1) * S + s];
  float qz = newxyz[((size_t)b * 3 + 2) * S + s];
  int* o = out + ((size_t)b * S + s) * K;
  int cnt = 0, first = 0;
  for (int i = 0; i < N && cnt < K; ++i) {
    float dx = X[i] - qx, dy = X[N + i] - qy, dz = X[2 * N + i] - qz;
    if (dx * dx + dy * dy + dz * dz <= r2) {
      if (cnt == 0) first = i;
      o[cnt++] = i;
    }
  }
  for (; cnt < K; ++cnt) o[cnt] = first;
}

// point-feature loaders: f32 plain or f16 PK2 (C even)
__device__ __forceinline__ float load_pt(const float* p, int b, int C, int c, int N, int id) {
  return p[((size_t)b * C + c) * N + id];
}
__device__ __forceinline__ float load_pt(const h16* p, int b, int C, int c, int N, int id) {
  return (float)p[pk2i(b, c, id, C >> 1, N)];
}

// grouped feats (B, 3+Cpts, S, K) -> f16 PK2 with Chalf = ceil((3+Cpts)/2)
template <typename TP>
__global__ void group_kernel(const float* __restrict__ xyz, const float* __restrict__ newxyz,
                             const TP* __restrict__ pts, const int* __restrict__ idx,
                             h16* __restrict__ out, int Cpts, int Chalf, int N, int S, int K,
                             size_t total) {
  size_t t = (size_t)blockIdx.x * blockDim.x + threadIdx.x;
  if (t >= total) return;
  int k = (int)(t % (size_t)K);
  size_t r = t / (size_t)K;
  int s = (int)(r % (size_t)S); r /= (size_t)S;
  int ch = (int)(r % (size_t)(3 + Cpts));
  int b = (int)(r / (size_t)(3 + Cpts));
  int id = idx[((size_t)b * S + s) * K + k];
  float v;
  if (ch < 3) v = xyz[((size_t)b * 3 + ch) * N + id] - newxyz[((size_t)b * 3 + ch) * S + s];
  else        v = load_pt(pts, b, Cpts, ch - 3, N, id);
  out[pk2i(b, ch, s * K + k, Chalf, S * K)] = (h16)v;
}

// max over K; in/out PK2, C even
__global__ void maxpool_kernel(const h16* __restrict__ in, h16* __restrict__ out,
                               int C, int S, int K, size_t total) {
  size_t t = (size_t)blockIdx.x * blockDim.x + threadIdx.x;
  if (t >= total) return;
  int s = (int)(t % (size_t)S);
  int c = (int)((t / (size_t)S) % (size_t)C);
  int b = (int)(t / ((size_t)S * C));
  const h16* ip = in + pk2i(b, c, s * K, C >> 1, S * K);
  float m = (float)ip[0];
  for (int k = 1; k < K; ++k) m = fmaxf(m, (float)ip[2 * k]);
  out[pk2i(b, c, s, C >> 1, S)] = (h16)m;
}

// ======================= misc elementwise ============================
// f32 plain src -> PK2 dst at channel offset
__global__ void copy_chF_kernel(const float* __restrict__ src, h16* __restrict__ dst,
                                int C, int ChalfD, int coff, int P, size_t total) {
  size_t t = (size_t)blockIdx.x * blockDim.x + threadIdx.x;
  if (t >= total) return;
  int p = (int)(t % (size_t)P);
  int c = (int)((t / (size_t)P) % (size_t)C);
  int b = (int)(t / ((size_t)C * P));
  dst[pk2i(b, coff + c, p, ChalfD, P)] = (h16)src[t];
}

// PK2 src (C even) -> PK2 dst at channel offset
__global__ void copy_chH_kernel(const h16* __restrict__ src, h16* __restrict__ dst,
                                int C, int ChalfD, int coff, int P, size_t total) {
  size_t t = (size_t)blockIdx.x * blockDim.x + threadIdx.x;
  if (t >= total) return;
  int p = (int)(t % (size_t)P);
  int c = (int)((t / (size_t)P) % (size_t)C);
  int b = (int)(t / ((size_t)C * P));
  dst[pk2i(b, coff + c, p, ChalfD, P)] = src[pk2i(b, c, p, C >> 1, P)];
}

// dst[coff+c] = r(PK2) * prev(f32 plain)
__global__ void mul_into_kernel(const h16* __restrict__ a, const float* __restrict__ bsrc,
                                h16* __restrict__ dst, int C, int ChalfD, int coff, int P,
                                size_t total) {
  size_t t = (size_t)blockIdx.x * blockDim.x + threadIdx.x;
  if (t >= total) return;
  int p = (int)(t % (size_t)P);
  int c = (int)((t / (size_t)P) % (size_t)C);
  int b = (int)(t / ((size_t)C * P));
  float av = (float)a[pk2i(b, c, p, C >> 1, P)];
  dst[pk2i(b, coff + c, p, ChalfD, P)] = (h16)(av * bsrc[t]);
}

// zero pad channels [c0, c1) of a PK2 buffer
__global__ void zero_ch_kernel(h16* __restrict__ dst, int c0, int c1, int Chalf, int P,
                               size_t total) {
  size_t t = (size_t)blockIdx.x * blockDim.x + threadIdx.x;
  if (t >= total) return;
  int nc = c1 - c0;
  int p  = (int)(t % (size_t)P);
  int ci = (int)((t / (size_t)P) % (size_t)nc);
  int b  = (int)(t / ((size_t)nc * P));
  dst[pk2i(b, c0 + ci, p, Chalf, P)] = (h16)0;
}

__global__ void fill_kernel(float* __restrict__ p, float v, size_t n) {
  size_t t = (size_t)blockIdx.x * blockDim.x + threadIdx.x;
  if (t < n) p[t] = v;
}

// 3-NN (or k<3 if M<3) with inverse-distance weights
__global__ void knn3_kernel(const float* __restrict__ xyz1, const float* __restrict__ xyz2,
                            int* __restrict__ kidx, float* __restrict__ kw,
                            int Nq, int M, int k, size_t total) {
  size_t t = (size_t)blockIdx.x * blockDim.x + threadIdx.x;
  if (t >= total) return;
  int b = (int)(t / (size_t)Nq), i = (int)(t % (size_t)Nq);
  float qx = xyz1[((size_t)b * 3 + 0) * Nq + i];
  float qy = xyz1[((size_t)b * 3 + 1) * Nq + i];
  float qz = xyz1[((size_t)b * 3 + 2) * Nq + i];
  float d0 = 1e30f, d1 = 1e30f, d2 = 1e30f;
  int i0 = 0, i1 = 0, i2 = 0;
  for (int j = 0; j < M; ++j) {
    float dx = xyz2[((size_t)b * 3 + 0) * M + j] - qx;
    float dy = xyz2[((size_t)b * 3 + 1) * M + j] - qy;
    float dz = xyz2[((size_t)b * 3 + 2) * M + j] - qz;
    float d = dx * dx + dy * dy + dz * dz;
    if (d < d0)      { d2 = d1; i2 = i1; d1 = d0; i1 = i0; d0 = d; i0 = j; }
    else if (d < d1) { d2 = d1; i2 = i1; d1 = d; i1 = j; }
    else if (d < d2) { d2 = d; i2 = j; }
  }
  float dd[3] = {d0, d1, d2};
  int   ii[3] = {i0, i1, i2};
  float rc[3] = {0.0f, 0.0f, 0.0f};
  float rsum = 0.0f;
  for (int j = 0; j < k; ++j) { rc[j] = 1.0f / fmaxf(dd[j], 1e-10f); rsum += rc[j]; }
  for (int j = 0; j < 3; ++j) {
    kidx[t * 3 + j] = (j < k) ? ii[j] : 0;
    kw[t * 3 + j]   = (j < k) ? rc[j] / rsum : 0.0f;
  }
}

// interpolate coarse PK2 features into PK2 concat buffer at coff=0
__global__ void interp_kernel(const h16* __restrict__ pts2, const int* __restrict__ kidx,
                              const float* __restrict__ kw, h16* __restrict__ dst,
                              int C2, int ChalfD, int M, int Nq, size_t total) {
  size_t t = (size_t)blockIdx.x * blockDim.x + threadIdx.x;
  if (t >= total) return;
  int i = (int)(t % (size_t)Nq);
  int c = (int)((t / (size_t)Nq) % (size_t)C2);
  int b = (int)(t / ((size_t)C2 * Nq));
  const int*   id = kidx + ((size_t)b * Nq + i) * 3;
  const float* ww = kw   + ((size_t)b * Nq + i) * 3;
  int Ch2 = C2 >> 1;
  float acc = (float)pts2[pk2i(b, c, id[0], Ch2, M)] * ww[0]
            + (float)pts2[pk2i(b, c, id[1], Ch2, M)] * ww[1]
            + (float)pts2[pk2i(b, c, id[2], Ch2, M)] * ww[2];
  dst[pk2i(b, c, i, ChalfD, Nq)] = (h16)acc;
}

// GRU combine: cur = (1-z)*cur + z*h_hat  (identical PK2 layouts -> linear)
__global__ void gru_combine_kernel(h16* __restrict__ cur, const h16* __restrict__ z,
                                   const h16* __restrict__ h, size_t n) {
  size_t t = (size_t)blockIdx.x * blockDim.x + threadIdx.x;
  if (t >= n) return;
  float zz = (float)z[t];
  cur[t] = (h16)((1.0f - zz) * (float)cur[t] + zz * (float)h[t]);
}

// delta = tanh(y)*0.1; out[0:n]=pc+delta; out[n:2n]=delta
__global__ void tanh_out_kernel(const float* __restrict__ y, const float* __restrict__ pc,
                                float* __restrict__ out, size_t n) {
  size_t t = (size_t)blockIdx.x * blockDim.x + threadIdx.x;
  if (t >= n) return;
  float d = tanhf(y[t]) * 0.1f;
  out[t] = pc[t] + d;
  out[n + t] = d;
}

// ========================= host orchestration ========================
struct ConvP { const float *w, *b, *g, *be; const h16* w16; int cin, cinp, cina, cout; };

static inline unsigned nblk(size_t n, unsigned bs) { return (unsigned)((n + bs - 1) / bs); }

static void conv_bn_act(hipStream_t st, const ConvP& p, const h16* X, float* Yf,
                        h16* Xout, int P, int act, float* sc, float* sh) {
  dim3 grid((unsigned)(P / 64), (unsigned)((p.cout + 15) / 16), (unsigned)BATCH);
  gemm_wmma_kernel<<<grid, 32, 0, st>>>(X, p.w16, p.b, Yf, p.cinp, p.cina, p.cout, P);
  if (p.g) {
    bn_stats_kernel<<<p.cout, 256, 0, st>>>(Yf, sc, sh, p.g, p.be, p.cout, P);
    size_t tot = (size_t)BATCH * p.cout * P;
    bn_apply_kernel<<<nblk(tot, 256), 256, 0, st>>>(Yf, Xout, sc, sh, p.cout, P, act, tot);
  }
}

static void unit_apply(hipStream_t st, const ConvP& ph, const ConvP& pr, const ConvP& pz,
                       h16* cur, const float* prev, int C, int P,
                       h16* cat, h16* zbuf, h16* tmp, float* Yf, float* sc, float* sh) {
  size_t n = (size_t)BATCH * C * P;
  copy_chH_kernel<<<nblk(n, 256), 256, 0, st>>>(cur,  cat, C, C, 0, P, n);
  copy_chF_kernel<<<nblk(n, 256), 256, 0, st>>>(prev, cat, C, C, C, P, n);
  conv_bn_act(st, pz, cat, Yf, zbuf, P, ACT_SIGMOID, sc, sh);  // z gate
  conv_bn_act(st, pr, cat, Yf, tmp,  P, ACT_SIGMOID, sc, sh);  // r gate
  // cat2 = [cur (already in place), r*prev]
  mul_into_kernel<<<nblk(n, 256), 256, 0, st>>>(tmp, prev, cat, C, C, C, P, n);
  conv_bn_act(st, ph, cat, Yf, tmp, P, ACT_RELU, sc, sh);      // h_hat
  gru_combine_kernel<<<nblk(n, 256), 256, 0, st>>>(cur, zbuf, tmp, n);
}

extern "C" void kernel_launch(void* const* d_in, const int* in_sizes, int n_in,
                              void* d_out, int out_size, void* d_ws, size_t ws_size,
                              hipStream_t stream) {
  (void)in_sizes; (void)n_in; (void)out_size; (void)ws_size;
  const float* pc      = (const float*)d_in[0];
  const float* prev_l0 = (const float*)d_in[1];
  const float* prev_l1 = (const float*)d_in[2];
  const float* prev_l2 = (const float*)d_in[3];
  const float* noise32 = (const float*)d_in[4];

  // params as JAX pytree leaves: dict keys sorted; conv dict => b, be, g, w
  int ix = 5;
  auto take = [&](int cin, int cout, bool bn) -> ConvP {
    ConvP p; p.cin = cin; p.cinp = (cin + 1) & ~1; p.cina = (cin + 31) & ~31;
    p.cout = cout; p.w16 = nullptr;
    p.b = (const float*)d_in[ix++];
    if (bn) { p.be = (const float*)d_in[ix++]; p.g = (const float*)d_in[ix++]; }
    else    { p.be = nullptr; p.g = nullptr; }
    p.w = (const float*)d_in[ix++];
    return p;
  };
  ConvP fin0 = take(160, 128, true), fin1 = take(128, 64, true), fin2 = take(64, 3, false);
  ConvP fp1a = take(134, 128, true), fp1b = take(128, 128, true), fp1c = take(128, 128, true);
  ConvP fp2a = take(384, 256, true), fp2b = take(256, 128, true);
  ConvP fp3a = take(1280, 256, true), fp3b = take(256, 256, true);
  ConvP sa1a = take(6, 64, true),  sa1b = take(64, 64, true),  sa1c = take(64, 128, true);
  ConvP sa2a = take(131, 128, true), sa2b = take(128, 128, true), sa2c = take(128, 256, true);
  ConvP sa3a = take(259, 256, true), sa3b = take(256, 512, true), sa3c = take(512, 1024, true);
  ConvP u1h = take(256, 128, true), u1r = take(256, 128, true), u1z = take(256, 128, true);
  ConvP u2h = take(256, 128, true), u2r = take(256, 128, true), u2z = take(256, 128, true);
  ConvP u3h = take(512, 256, true), u3r = take(512, 256, true), u3z = take(512, 256, true);

  // ---- workspace bump allocator ----
  char* wsp = (char*)d_ws;
  size_t off = 0;
  auto alloc = [&](size_t bytes) -> char* {
    char* p = wsp + off;
    off += (bytes + 255) & ~(size_t)255;
    return p;
  };
  float* YF = (float*)alloc((size_t)BATCH * 128 * 16384 * 4);  // f32 pre-BN GEMM out
  h16* H0 = (h16*)alloc((size_t)BATCH * 128 * 16384 * 2);      // f16 act ping (PK2)
  h16* H1 = (h16*)alloc((size_t)BATCH * 66 * 16384 * 2);       // f16 act pong (PK2)
  h16* ZB = (h16*)alloc((size_t)BATCH * 128 * 2048 * 2);       // z gate
  float* l1_xyz = (float*)alloc((size_t)BATCH * 3 * 512 * 4);
  float* l2_xyz = (float*)alloc((size_t)BATCH * 3 * 128 * 4);
  float* l3_xyz = (float*)alloc((size_t)BATCH * 3 * 4);
  int*   fpsidx = (int*)alloc((size_t)BATCH * 512 * 4);
  int*   ballidx = (int*)alloc((size_t)BATCH * 512 * 32 * 4);
  h16* l1p  = (h16*)alloc((size_t)BATCH * 128 * 512 * 2);
  h16* l2p  = (h16*)alloc((size_t)BATCH * 256 * 128 * 2);
  h16* l3p  = (h16*)alloc((size_t)BATCH * 1024 * 2);
  h16* l2pf = (h16*)alloc((size_t)BATCH * 256 * 128 * 2);
  h16* l1pf = (h16*)alloc((size_t)BATCH * 128 * 512 * 2);
  h16* l0pf = (h16*)alloc((size_t)BATCH * 128 * 2048 * 2);
  int*   kidx = (int*)alloc((size_t)BATCH * 2048 * 3 * 4);
  float* kw   = (float*)alloc((size_t)BATCH * 2048 * 3 * 4);
  h16* p1buf = (h16*)alloc((size_t)BATCH * 6 * 2048 * 2);
  float* sc = (float*)alloc(1024 * 4);
  float* sh = (float*)alloc(1024 * 4);
  h16* wpool = (h16*)alloc((size_t)2400000 * 2);               // f16 padded weights

  // pre-convert all weights to f16 with zero-padded 32-aligned rows
  {
    ConvP* all[] = {&fin0,&fin1,&fin2,&fp1a,&fp1b,&fp1c,&fp2a,&fp2b,&fp3a,&fp3b,
                    &sa1a,&sa1b,&sa1c,&sa2a,&sa2b,&sa2c,&sa3a,&sa3b,&sa3c,
                    &u1h,&u1r,&u1z,&u2h,&u2r,&u2z,&u3h,&u3r,&u3z};
    size_t woff = 0;
    for (ConvP* p : all) {
      size_t n = (size_t)p->cina * p->cout;
      h16* dst = wpool + woff; woff += n;
      cvt16pad_kernel<<<nblk(n, 256), 256, 0, stream>>>(p->w, dst, p->cin, p->cina, n);
      p->w16 = dst;
    }
  }

  // ===================== SA1: N=2048 -> S=512, K=32, r=0.2 =====================
  fps_kernel<<<BATCH, 256, 0, stream>>>(pc, fpsidx, NPTS, 512);
  { size_t t = (size_t)BATCH * 3 * 512;
    gather_xyz_kernel<<<nblk(t, 256), 256, 0, stream>>>(pc, fpsidx, l1_xyz, NPTS, 512, t); }
  { size_t t = (size_t)BATCH * 512;
    ball_query_kernel<<<nblk(t, 128), 128, 0, stream>>>(pc, l1_xyz, ballidx, NPTS, 512, 32, 0.04f, t); }
  { size_t t = (size_t)BATCH * 6 * 512 * 32;
    group_kernel<float><<<nblk(t, 256), 256, 0, stream>>>(pc, l1_xyz, pc, ballidx, H1, 3, 3, NPTS, 512, 32, t); }
  conv_bn_act(stream, sa1a, H1, YF, H0, 512 * 32, ACT_RELU, sc, sh);
  conv_bn_act(stream, sa1b, H0, YF, H1, 512 * 32, ACT_RELU, sc, sh);
  conv_bn_act(stream, sa1c, H1, YF, H0, 512 * 32, ACT_RELU, sc, sh);
  { size_t t = (size_t)BATCH * 128 * 512;
    maxpool_kernel<<<nblk(t, 256), 256, 0, stream>>>(H0, l1p, 128, 512, 32, t); }

  // ===================== SA2: N=512 -> S=128, K=32, r=0.4 =====================
  fps_kernel<<<BATCH, 256, 0, stream>>>(l1_xyz, fpsidx, 512, 128);
  { size_t t = (size_t)BATCH * 3 * 128;
    gather_xyz_kernel<<<nblk(t, 256), 256, 0, stream>>>(l1_xyz, fpsidx, l2_xyz, 512, 128, t); }
  { size_t t = (size_t)BATCH * 128;
    ball_query_kernel<<<nblk(t, 128), 128, 0, stream>>>(l1_xyz, l2_xyz, ballidx, 512, 128, 32, 0.16f, t); }
  { size_t t = (size_t)BATCH * 131 * 128 * 32;   // 131 ch -> pad to 132 (Chalf 66)
    group_kernel<h16><<<nblk(t, 256), 256, 0, stream>>>(l1_xyz, l2_xyz, l1p, ballidx, H1, 128, 66, 512, 128, 32, t); }
  { size_t t = (size_t)BATCH * 1 * 4096;
    zero_ch_kernel<<<nblk(t, 256), 256, 0, stream>>>(H1, 131, 132, 66, 4096, t); }
  conv_bn_act(stream, sa2a, H1, YF, H0, 128 * 32, ACT_RELU, sc, sh);
  conv_bn_act(stream, sa2b, H0, YF, H1, 128 * 32, ACT_RELU, sc, sh);
  conv_bn_act(stream, sa2c, H1, YF, H0, 128 * 32, ACT_RELU, sc, sh);
  { size_t t = (size_t)BATCH * 256 * 128;
    maxpool_kernel<<<nblk(t, 256), 256, 0, stream>>>(H0, l2p, 256, 128, 32, t); }

  // ===================== SA3: group_all over 128 points (259 -> pad 260) ======
  { size_t t = (size_t)BATCH * 3 * 128;
    copy_chF_kernel<<<nblk(t, 256), 256, 0, stream>>>(l2_xyz, H1, 3, 130, 0, 128, t); }
  { size_t t = (size_t)BATCH * 256 * 128;
    copy_chH_kernel<<<nblk(t, 256), 256, 0, stream>>>(l2p, H1, 256, 130, 3, 128, t); }
  { size_t t = (size_t)BATCH * 1 * 128;
    zero_ch_kernel<<<nblk(t, 256), 256, 0, stream>>>(H1, 259, 260, 130, 128, t); }
  conv_bn_act(stream, sa3a, H1, YF, H0, 128, ACT_RELU, sc, sh);
  conv_bn_act(stream, sa3b, H0, YF, H1, 128, ACT_RELU, sc, sh);
  conv_bn_act(stream, sa3c, H1, YF, H0, 128, ACT_RELU, sc, sh);
  { size_t t = (size_t)BATCH * 1024;
    maxpool_kernel<<<nblk(t, 256), 256, 0, stream>>>(H0, l3p, 1024, 1, 128, t); }
  fill_kernel<<<1, 128, 0, stream>>>(l3_xyz, 0.0f, (size_t)BATCH * 3);

  // ===================== FP3 (M=1 -> k=1 broadcast) =====================
  { size_t t = (size_t)BATCH * 128;
    knn3_kernel<<<nblk(t, 128), 128, 0, stream>>>(l2_xyz, l3_xyz, kidx, kw, 128, 1, 1, t); }
  { size_t t = (size_t)BATCH * 1024 * 128;
    interp_kernel<<<nblk(t, 256), 256, 0, stream>>>(l3p, kidx, kw, H1, 1024, 640, 1, 128, t); }
  { size_t t = (size_t)BATCH * 256 * 128;
    copy_chH_kernel<<<nblk(t, 256), 256, 0, stream>>>(l2p, H1, 256, 640, 1024, 128, t); }
  conv_bn_act(stream, fp3a, H1, YF, H0, 128, ACT_RELU, sc, sh);
  conv_bn_act(stream, fp3b, H0, YF, l2pf, 128, ACT_RELU, sc, sh);
  unit_apply(stream, u3h, u3r, u3z, l2pf, prev_l2, 256, 128, H1, ZB, H0, YF, sc, sh);

  // ===================== FP2 =====================
  { size_t t = (size_t)BATCH * 512;
    knn3_kernel<<<nblk(t, 128), 128, 0, stream>>>(l1_xyz, l2_xyz, kidx, kw, 512, 128, 3, t); }
  { size_t t = (size_t)BATCH * 256 * 512;
    interp_kernel<<<nblk(t, 256), 256, 0, stream>>>(l2pf, kidx, kw, H1, 256, 192, 128, 512, t); }
  { size_t t = (size_t)BATCH * 128 * 512;
    copy_chH_kernel<<<nblk(t, 256), 256, 0, stream>>>(l1p, H1, 128, 192, 256, 512, t); }
  conv_bn_act(stream, fp2a, H1, YF, H0, 512, ACT_RELU, sc, sh);
  conv_bn_act(stream, fp2b, H0, YF, l1pf, 512, ACT_RELU, sc, sh);
  unit_apply(stream, u2h, u2r, u2z, l1pf, prev_l1, 128, 512, H1, ZB, H0, YF, sc, sh);

  // ===================== FP1 =====================
  { size_t t = (size_t)BATCH * 3 * NPTS;
    copy_chF_kernel<<<nblk(t, 256), 256, 0, stream>>>(pc, p1buf, 3, 3, 0, NPTS, t);
    copy_chF_kernel<<<nblk(t, 256), 256, 0, stream>>>(pc, p1buf, 3, 3, 3, NPTS, t); }
  { size_t t = (size_t)BATCH * NPTS;
    knn3_kernel<<<nblk(t, 128), 128, 0, stream>>>(pc, l1_xyz, kidx, kw, NPTS, 512, 3, t); }
  { size_t t = (size_t)BATCH * 128 * NPTS;
    interp_kernel<<<nblk(t, 256), 256, 0, stream>>>(l1pf, kidx, kw, H1, 128, 67, 512, NPTS, t); }
  { size_t t = (size_t)BATCH * 6 * NPTS;
    copy_chH_kernel<<<nblk(t, 256), 256, 0, stream>>>(p1buf, H1, 6, 67, 128, NPTS, t); }
  conv_bn_act(stream, fp1a, H1, YF, H0, NPTS, ACT_RELU, sc, sh);
  conv_bn_act(stream, fp1b, H0, YF, H1, NPTS, ACT_RELU, sc, sh);
  conv_bn_act(stream, fp1c, H1, YF, l0pf, NPTS, ACT_RELU, sc, sh);
  unit_apply(stream, u1h, u1r, u1z, l0pf, prev_l0, 128, NPTS, H1, ZB, H0, YF, sc, sh);

  // ===================== FINAL =====================
  { size_t t = (size_t)BATCH * 128 * NPTS;
    copy_chH_kernel<<<nblk(t, 256), 256, 0, stream>>>(l0pf, H1, 128, 80, 0, NPTS, t); }
  { size_t t = (size_t)BATCH * 32 * NPTS;
    copy_chF_kernel<<<nblk(t, 256), 256, 0, stream>>>(noise32, H1, 32, 80, 128, NPTS, t); }
  conv_bn_act(stream, fin0, H1, YF, H0, NPTS, ACT_RELU, sc, sh);
  conv_bn_act(stream, fin1, H0, YF, H1, NPTS, ACT_RELU, sc, sh);
  conv_bn_act(stream, fin2, H1, YF, nullptr, NPTS, ACT_NONE, sc, sh);  // bias only
  { size_t n = (size_t)BATCH * 3 * NPTS;
    tanh_out_kernel<<<nblk(n, 256), 256, 0, stream>>>(YF, pc, (float*)d_out, n); }
}
